// Encoder_80994493268501
// MI455X (gfx1250) — matrix operand order repaired and checked by reading, not compiled, not run on hardware
//
#include <hip/hip_runtime.h>

typedef __attribute__((ext_vector_type(16))) _Float16 v16h;
typedef __attribute__((ext_vector_type(8)))  _Float16 v8h;
typedef __attribute__((ext_vector_type(8)))  float    v8f;
typedef __attribute__((ext_vector_type(4)))  float    v4f;

#define NIN  128
#define NOUT 256

// ---------------- graph preprocessing ----------------

__global__ void k_zero_u32(unsigned* __restrict__ p, int n) {
  int i = blockIdx.x * blockDim.x + threadIdx.x;
  if (i < n) p[i] = 0u;
}

// degree over target index (col = ei[1])
__global__ void k_degree(const long long* __restrict__ ei, unsigned* __restrict__ deg, int E) {
  int e = blockIdx.x * blockDim.x + threadIdx.x;
  if (e < E) {
    long long c = ei[(size_t)E + e];
    atomicAdd(&deg[(size_t)c], 1u);
  }
}

// dinv = rsqrt(deg + 1)   (+1 = self loop, so deg > 0 always)
__global__ void k_dinv(const unsigned* __restrict__ deg, float* __restrict__ dinv, int n) {
  int i = blockIdx.x * blockDim.x + threadIdx.x;
  if (i < n) dinv[i] = rsqrtf((float)(deg[i] + 1u));
}

// agg[i][:] = dinv[i]^2 * x[i][:]   (self-loop contribution)
__global__ void k_init_agg(const float* __restrict__ x, const float* __restrict__ dinv,
                           float* __restrict__ agg, int total) {
  int i = blockIdx.x * blockDim.x + threadIdx.x;
  if (i < total) {
    int node = i >> 7;  // / NIN (=128)
    float di = dinv[node];
    agg[i] = di * di * x[i];
  }
}

// one wave32 per edge; each lane owns 4 consecutive features
__global__ __launch_bounds__(256) void k_scatter(const long long* __restrict__ ei,
                                                 const float* __restrict__ x,
                                                 const float* __restrict__ dinv,
                                                 float* __restrict__ agg, int E) {
  int lane = threadIdx.x & 31;   // wave32 on gfx1250
  int wave = threadIdx.x >> 5;
  int e = blockIdx.x * 8 + wave;
  if (e >= E) return;
  long long r = ei[e];               // source
  long long c = ei[(size_t)E + e];   // target
  float norm = dinv[(size_t)r] * dinv[(size_t)c];
  v4f xv = *(const v4f*)(x + (size_t)r * NIN + lane * 4);
  float* dst = agg + (size_t)c * NIN + lane * 4;
  atomicAdd(dst + 0, norm * xv.x);
  atomicAdd(dst + 1, norm * xv.y);
  atomicAdd(dst + 2, norm * xv.z);
  atomicAdd(dst + 3, norm * xv.w);
}

__global__ void k_f32_to_f16(const float* __restrict__ s, _Float16* __restrict__ d, int n) {
  int i = blockIdx.x * blockDim.x + threadIdx.x;
  if (i < n) d[i] = (_Float16)s[i];
}

// W: [K][NOUT] f32 row-major  ->  WT: [NOUT][K] f16
__global__ void k_transpose_w(const float* __restrict__ W, _Float16* __restrict__ WT, int K) {
  int i = blockIdx.x * blockDim.x + threadIdx.x;
  if (i < K * NOUT) {
    int k = i / NOUT, n2 = i - k * NOUT;
    WT[(size_t)n2 * K + k] = (_Float16)W[i];
  }
}

// ---------------- WMMA GEMM + ReLU ----------------
// C[M][256] = relu(A[M][K] * B[K][256]); BT given as [256][K] f16.
// 8 waves/block: wave (rgrp,cgrp) computes rows rowBase..+15, cols cgrp*64..+63
// via 4 v_wmma_f32_16x16x32_f16 accumulators. Weight K-slices stream into a
// double-buffered LDS tile with async Global->LDS copies (ASYNCcnt).
// ISA: INST_OFFSET applies to BOTH the LDS dest and the global src address,
// so per-chunk addressing is a pure immediate (no VALU address math).
__global__ __launch_bounds__(256) void gcn_gemm_relu(
    const _Float16* __restrict__ A,   // [M][K]
    const _Float16* __restrict__ BT,  // [NOUT][K]
    int M, int K,
    float*    __restrict__ outF,      // [M][NOUT] or nullptr
    _Float16* __restrict__ outH)      // [M][NOUT] or nullptr
{
  __shared__ _Float16 sBT[2][NOUT * 64];   // 2 x 32 KB K-slices of weights

  const int tid  = threadIdx.x;
  const int lane = tid & 31;
  const int wave = tid >> 5;
  const int rgrp = wave >> 2;          // 0..1
  const int cgrp = wave & 3;           // 0..3
  const int rowBase = blockIdx.x * 32 + rgrp * 16;
  const int colBase = cgrp * 64;
  const int l15  = lane & 15;
  const int hiH  = (lane >= 16) ? 1 : 0;

  // clamp row for loads (stores are guarded); keeps EXEC all-1s for WMMA
  int mrow = rowBase + l15;
  int arow = (mrow < M) ? mrow : (M - 1);
  const _Float16* Arow = A + (size_t)arow * K + (hiH ? 16 : 0);

  // per-thread async copy job: 128 B of BT row `tid` for one 64-wide K slice
  const _Float16* gsrcBase = BT + (size_t)tid * K;
  unsigned ldsBase0 = (unsigned)(size_t)&sBT[0][tid * 64];  // addrspace(3) offset
  unsigned ldsBase1 = (unsigned)(size_t)&sBT[1][tid * 64];

  auto issue_slice_copy = [&](int slice, int buf) {
    unsigned lb = buf ? ldsBase1 : ldsBase0;
    const char* g = (const char*)(gsrcBase + slice * 64);
    #pragma unroll
    for (int j = 0; j < 8; ++j) {
      asm volatile("global_load_async_to_lds_b128 %0, %1, off offset:%c2"
                   :: "v"(lb), "v"(g), "i"(j * 16)
                   : "memory");
    }
  };

  v8f acc[4];
  acc[0] = (v8f){}; acc[1] = (v8f){}; acc[2] = (v8f){}; acc[3] = (v8f){};

  const int nslices = K >> 6;           // K / 64
  issue_slice_copy(0, 0);
  asm volatile("s_wait_asynccnt 0x0" ::: "memory");
  __syncthreads();

  for (int sl = 0; sl < nslices; ++sl) {
    const int buf = sl & 1;
    if (sl + 1 < nslices)
      issue_slice_copy(sl + 1, buf ^ 1);            // overlap with compute

    const int kk0 = sl * 64;
    const _Float16* sb = &sBT[buf][0];

    #pragma unroll
    for (int s = 0; s < 2; ++s) {
      const int kk = kk0 + s * 32;
      // A fragment (16x32 f16): lanes 0-15 rows M=l, K 0..15; lanes 16-31 K 16..31
      v8h a0 = *(const v8h*)(Arow + kk);
      v8h a1 = *(const v8h*)(Arow + kk + 8);
      v16h a = __builtin_shufflevector(a0, a1, 0,1,2,3,4,5,6,7,8,9,10,11,12,13,14,15);

      // B fragments (32x16 f16): lane holds col l15, 16 contiguous K (+16 for hi half)
      const int klocal = s * 32 + (hiH ? 16 : 0);
      const _Float16* bcol = sb + (size_t)(colBase + l15) * 64 + klocal;
      #pragma unroll
      for (int t = 0; t < 4; ++t) {
        v16h b = *(const v16h*)(bcol + t * 16 * 64);
        acc[t] = __builtin_amdgcn_wmma_f32_16x16x32_f16(
            false, a, false, b, (short)0, acc[t], false, false);
      }
    }

    asm volatile("s_wait_asynccnt 0x0" ::: "memory");  // next slice landed in LDS
    __syncthreads();
  }

  // C/D layout: VGPR r, lanes 0-15 -> M=r, lanes 16-31 -> M=r+8; N = lane&15
  #pragma unroll
  for (int t = 0; t < 4; ++t) {
    int col = colBase + t * 16 + l15;
    #pragma unroll
    for (int r = 0; r < 8; ++r) {
      int row = rowBase + r + (hiH ? 8 : 0);
      if (row < M) {
        float v = acc[t][r];
        v = v > 0.0f ? v : 0.0f;                 // ReLU
        if (outF) outF[(size_t)row * NOUT + col] = v;
        if (outH) outH[(size_t)row * NOUT + col] = (_Float16)v;
      }
    }
  }
}

// ---------------- driver ----------------

extern "C" void kernel_launch(void* const* d_in, const int* in_sizes, int n_in,
                              void* d_out, int out_size, void* d_ws, size_t ws_size,
                              hipStream_t stream) {
  const float*     x  = (const float*)d_in[0];
  const float*     W0 = (const float*)d_in[1];   // [128][256]
  const float*     W1 = (const float*)d_in[2];   // [256][256]
  const long long* ei = (const long long*)d_in[3];

  const int N = in_sizes[0] / NIN;
  const int E = in_sizes[3] / 2;

  char* ws = (char*)d_ws;
  size_t off = 0;
  auto carve = [&](size_t bytes) -> char* {
    char* p = ws + off;
    off += (bytes + 255) & ~(size_t)255;
    return p;
  };
  unsigned* degU  = (unsigned*)carve((size_t)N * 4);
  float*    dinv  = (float*)   carve((size_t)N * 4);
  float*    agg   = (float*)   carve((size_t)N * NIN * 4);
  _Float16* aggH  = (_Float16*)carve((size_t)N * NIN * 2);
  _Float16* W0T   = (_Float16*)carve((size_t)NOUT * NIN * 2);
  _Float16* W1T   = (_Float16*)carve((size_t)NOUT * NOUT * 2);
  _Float16* H1    = (_Float16*)carve((size_t)N * NOUT * 2);
  (void)ws_size;

  const int B = 256;
  k_zero_u32   <<<(N + B - 1) / B, B, 0, stream>>>(degU, N);
  k_degree     <<<(E + B - 1) / B, B, 0, stream>>>(ei, degU, E);
  k_dinv       <<<(N + B - 1) / B, B, 0, stream>>>(degU, dinv, N);
  k_init_agg   <<<(N * NIN + B - 1) / B, B, 0, stream>>>(x, dinv, agg, N * NIN);
  k_scatter    <<<(E + 7) / 8, B, 0, stream>>>(ei, x, dinv, agg, E);
  k_f32_to_f16 <<<(N * NIN + B - 1) / B, B, 0, stream>>>(agg, aggH, N * NIN);
  k_transpose_w<<<(NIN * NOUT + B - 1) / B, B, 0, stream>>>(W0, W0T, NIN);
  k_transpose_w<<<(NOUT * NOUT + B - 1) / B, B, 0, stream>>>(W1, W1T, NOUT);

  // layer 1: H1 = relu(agg @ W0), f16 output feeds layer 2
  gcn_gemm_relu<<<(N + 31) / 32, B, 0, stream>>>(aggH, W0T, N, NIN, nullptr, H1);
  // layer 2: out = relu(H1 @ W1), f32 output
  gcn_gemm_relu<<<(N + 31) / 32, B, 0, stream>>>(H1, W1T, N, NOUT, (float*)d_out, nullptr);
}
